// Atom91SeqDecoder_47236050321508
// MI455X (gfx1250) — matrix operand order repaired and checked by reading, not compiled, not run on hardware
//
#include <hip/hip_runtime.h>
#include <math.h>

#define NN 8192
#define KK 30
#define NKE (NN*KK)

typedef __attribute__((ext_vector_type(16))) _Float16 v16h;
typedef __attribute__((ext_vector_type(8)))  _Float16 v8h;
typedef __attribute__((ext_vector_type(8)))  float    v8f;

#define GF_RELU 1
#define GF_ACC  2

__device__ __forceinline__ float wave_sum(float v) {
#pragma unroll
  for (int off = 16; off > 0; off >>= 1) v += __shfl_xor(v, off, 32);
  return v;
}

// ---------------------------------------------------------------------------
// Weight transpose + f16 convert: BT[n][k] = B[k][n] for n < Nb, else 0.
// BT is (Npad x Ka) row-major f16, Npad = ceil16(Nb). Columns of padded rows
// are zero-filled so the GEMM needs no masking on the multiply side.
// ---------------------------------------------------------------------------
__global__ void wtr_kernel(const float* __restrict__ B, int ldb, int Ka, int Nb,
                           int Npad, _Float16* __restrict__ BT) {
  int i = blockIdx.x*blockDim.x + threadIdx.x;
  if (i >= Npad * Ka) return;
  int k = i % Ka; int n = i / Ka;
  BT[i] = (n < Nb) ? (_Float16)B[(size_t)k*ldb + n] : (_Float16)0.f;
}

// ---------------------------------------------------------------------------
// Generic WMMA GEMM: C[M,Nb] = op(A[M,Ka] @ B[Ka,Nb] + bias), fp32 A/C,
// pre-transposed f16 B. One wave per 16x16 tile. Ka % 32 == 0, M % 16 == 0.
// B fragment = 16 contiguous halves -> two b128 loads, branch-free.
// ---------------------------------------------------------------------------
__global__ void wmma_gemm_kernel(const float* __restrict__ A, int lda,
                                 const _Float16* __restrict__ BT,
                                 float* __restrict__ C, int ldc,
                                 int M, int Ka, int Nb,
                                 const float* __restrict__ bias, int flags) {
  int ntiles = (Nb + 15) >> 4;
  int tile = blockIdx.x * (blockDim.x >> 5) + (threadIdx.x >> 5);
  int mt = tile / ntiles, nt = tile % ntiles;
  if (mt * 16 >= M) return;
  int lane = threadIdx.x & 31;
  int mrow = (mt << 4) + (lane & 15);
  int ncol = (nt << 4) + (lane & 15);
  const _Float16* Br = BT + (size_t)ncol * Ka + (lane & 16);
  v8f acc = {};
  for (int k0 = 0; k0 < Ka; k0 += 32) {
    const float* Ar = A + (size_t)mrow * lda + k0;
    if (k0 + 32 < Ka) __builtin_prefetch(Ar + 32, 0, 1);
    v16h a;
#pragma unroll
    for (int h = 0; h < 16; ++h) {
      int ka = h + (lane & 16);  // contiguous k = 16*(lane/16) + h, matches B fragment
      a[h] = (_Float16)Ar[ka];
    }
    const v8h* Bp = (const v8h*)(Br + k0);                     // 32B-aligned by construction
    v8h b0 = Bp[0], b1 = Bp[1];
    v16h b = __builtin_shufflevector(b0, b1, 0,1,2,3,4,5,6,7,8,9,10,11,12,13,14,15);
    acc = __builtin_amdgcn_wmma_f32_16x16x32_f16(false, a, false, b, (short)0, acc, false, false);
  }
  if (ncol < Nb) {
    int rbase = (mt << 4) + ((lane & 16) >> 1);
    float bv = bias ? bias[ncol] : 0.f;
#pragma unroll
    for (int j = 0; j < 8; ++j) {
      int row = rbase + j;
      float v = acc[j] + bv;
      if (flags & GF_RELU) v = fmaxf(v, 0.f);
      size_t off = (size_t)row * ldc + ncol;
      if (flags & GF_ACC) v += C[off];
      C[off] = v;
    }
  }
}

// ---------------------------------------------------------------------------
// Brute-force KNN: one wave per query node. Per-lane top-30 lists in LDS,
// merged with wave32 shfl arg-min rounds.
// ---------------------------------------------------------------------------
#define KW 4
__global__ void knn_kernel(const float* __restrict__ bb, const int* __restrict__ batch,
                           const unsigned char* __restrict__ xmask, int* __restrict__ nbr) {
  __shared__ float sd[KW * 32 * KK];
  __shared__ int   si[KW * 32 * KK];
  int wave = threadIdx.x >> 5, lane = threadIdx.x & 31;
  int n = blockIdx.x * KW + wave;
  if (n >= NN) return;
  float xi0 = bb[n*12+3], xi1 = bb[n*12+4], xi2 = bb[n*12+5];
  int bi = batch[n];
  float* md = sd + (wave*32 + lane) * KK;
  int*   mi = si + (wave*32 + lane) * KK;
#pragma unroll
  for (int t = 0; t < KK; ++t) { md[t] = 3.0e38f; mi[t] = 0; }
  float curmax = 3.0e38f; int amax = 0;
  for (int j = lane; j < NN; j += 32) {
    float dx = xi0 - bb[j*12+3];
    float dy = xi1 - bb[j*12+4];
    float dz = xi2 - bb[j*12+5];
    float d = dx*dx + dy*dy + dz*dz;
    if (j == n || batch[j] != bi || xmask[j]) d = 2.0e38f;
    if (d < curmax) {
      md[amax] = d; mi[amax] = j;
      curmax = md[0]; amax = 0;
      for (int t = 1; t < KK; ++t) if (md[t] > curmax) { curmax = md[t]; amax = t; }
    }
  }
  for (int t = 0; t < KK; ++t) {
    float bd = md[0]; int bs = 0;
    for (int s2 = 1; s2 < KK; ++s2) if (md[s2] < bd) { bd = md[s2]; bs = s2; }
    int bidx = mi[bs];
    int bl = lane;
#pragma unroll
    for (int off = 16; off > 0; off >>= 1) {
      float od = __shfl_xor(bd, off, 32);
      int oi = __shfl_xor(bidx, off, 32);
      int ol = __shfl_xor(bl, off, 32);
      int os = __shfl_xor(bs, off, 32);
      if (od < bd || (od == bd && ol < bl)) { bd = od; bidx = oi; bl = ol; bs = os; }
    }
    if (lane == bl) md[bs] = 3.9e38f;   // remove winner
    if (lane == 0) nbr[n*KK + t] = bidx;
  }
}

// ---------------------------------------------------------------------------
// Edge features: 16 RBF + 8 cos + 8 sin of relative sequence position.
// ---------------------------------------------------------------------------
__global__ void edge_feat_kernel(const float* __restrict__ bb, const int* __restrict__ nbr,
                                 float* __restrict__ edge) {
  int e = blockIdx.x*blockDim.x + threadIdx.x;
  if (e >= NKE) return;
  int n = e / KK; int j = nbr[e];
  float dx = bb[n*12+3]-bb[j*12+3];
  float dy = bb[n*12+4]-bb[j*12+4];
  float dz = bb[n*12+5]-bb[j*12+5];
  float dist = sqrtf(dx*dx+dy*dy+dz*dz + 1e-8f);
  const float sig = 1.25f;
#pragma unroll
  for (int i = 0; i < 16; ++i) {
    float mu = (float)i * (20.f/15.f);
    float t = (dist - mu) / sig;
    edge[(size_t)e*32 + i] = expf(-t*t);
  }
  float rel = (float)(j - n);
  const float lf = -logf(10000.f) / 16.f;
#pragma unroll
  for (int f = 0; f < 8; ++f) {
    float a = rel * expf((float)(2*f) * lf);
    edge[(size_t)e*32 + 16 + f] = cosf(a);
    edge[(size_t)e*32 + 24 + f] = sinf(a);
  }
}

// ---------------------------------------------------------------------------
// Backbone geometric embedding: dihedrals (row 0) + l1 vectors (rows 1..3).
// ---------------------------------------------------------------------------
__device__ __forceinline__ void cross3(const float* a, const float* b, float* o) {
  o[0]=a[1]*b[2]-a[2]*b[1]; o[1]=a[2]*b[0]-a[0]*b[2]; o[2]=a[0]*b[1]-a[1]*b[0];
}
__device__ __forceinline__ void norm3(float* v) {
  float inv = 1.0f / sqrtf(v[0]*v[0]+v[1]*v[1]+v[2]*v[2] + 1e-8f);
  v[0]*=inv; v[1]*=inv; v[2]*=inv;
}
__device__ __forceinline__ float dot3(const float* a, const float* b) {
  return a[0]*b[0]+a[1]*b[1]+a[2]*b[2];
}

__global__ void bbemb_kernel(const float* __restrict__ bb, float* __restrict__ be) {
  int n = blockIdx.x*blockDim.x + threadIdx.x;
  if (n >= NN) return;
  float D[3];
  const long NA = 3L*NN;
  for (int t = 0; t < 3; ++t) {
    long i = 3L*n + t - 1;
    if (i < 0 || i >= NA - 3) { D[t] = 0.f; continue; }
    float P[4][3];
    for (int q = 0; q < 4; ++q) {
      long a = i + q; int rn = (int)(a/3), ra = (int)(a%3);
      P[q][0]=bb[rn*12+ra*3+0]; P[q][1]=bb[rn*12+ra*3+1]; P[q][2]=bb[rn*12+ra*3+2];
    }
    float u2[3],u1[3],u0[3];
    for (int d=0; d<3; ++d) { u2[d]=P[1][d]-P[0][d]; u1[d]=P[2][d]-P[1][d]; u0[d]=P[3][d]-P[2][d]; }
    norm3(u2); norm3(u1); norm3(u0);
    float n2[3], n1v[3];
    cross3(u2,u1,n2); norm3(n2);
    cross3(u1,u0,n1v); norm3(n1v);
    float c = dot3(n2,n1v);
    c = fminf(fmaxf(c, -1.f+1e-7f), 1.f-1e-7f);
    float s = dot3(u2,n1v);
    float sg = (s > 0.f) ? 1.f : ((s < 0.f) ? -1.f : 0.f);
    D[t] = sg * acosf(c);
  }
  float* r0 = be + (size_t)n*28;
  r0[0]=cosf(D[0]); r0[1]=cosf(D[1]); r0[2]=cosf(D[2]);
  r0[3]=sinf(D[0]); r0[4]=sinf(D[1]); r0[5]=sinf(D[2]); r0[6]=0.f;
  float X[3] = {bb[n*12+3], bb[n*12+4], bb[n*12+5]};
  float l1[7][3];
  for (int c=0;c<4;++c) for (int d=0;d<3;++d) l1[c][d] = bb[n*12+c*3+d] - X[d];
  if (n < NN-1) { for (int d=0;d<3;++d) l1[4][d] = bb[(n+1)*12+3+d]-X[d]; norm3(l1[4]); }
  else          { l1[4][0]=l1[4][1]=l1[4][2]=0.f; }
  if (n > 0)    { for (int d=0;d<3;++d) l1[5][d] = bb[(n-1)*12+3+d]-X[d]; norm3(l1[5]); }
  else          { l1[5][0]=l1[5][1]=l1[5][2]=0.f; }
  float bv[3], cv[3], av[3];
  for (int d=0;d<3;++d) { bv[d]=bb[n*12+3+d]-bb[n*12+0+d]; cv[d]=bb[n*12+6+d]-bb[n*12+3+d]; }
  cross3(bv,cv,av);
  for (int d=0;d<3;++d) l1[6][d] = -0.58273431f*av[d] + 0.56802827f*bv[d] - 0.54067466f*cv[d];
  for (int d=0; d<3; ++d)
    for (int c=0; c<7; ++c)
      be[(size_t)n*28 + (1+d)*7 + c] = l1[c][d];
}

// g[n,m,c] = bb_emb[n,m,:] @ (m==0 ? W0 : W1)   (inner dim 7)
__global__ void gemb_kernel(const float* __restrict__ be, const float* __restrict__ W0,
                            const float* __restrict__ W1, float* __restrict__ g) {
  int i = blockIdx.x*blockDim.x + threadIdx.x;
  if (i >= NN*128) return;
  int c = i & 31; int rr = i >> 5; int m = rr & 3; int n = rr >> 2;
  const float* W = (m == 0) ? W0 : W1;
  float s = 0.f;
#pragma unroll
  for (int j = 0; j < 7; ++j) s += be[(size_t)n*28 + m*7 + j] * W[j*32 + c];
  g[i] = s;
}

// node[n,m,c] = mean_k g[nbr[n,k],m,c] * r[n,k,c]   (rotations cancel: Rt R = I)
__global__ void proj_reduce_kernel(const float* __restrict__ g, const float* __restrict__ r,
                                   const int* __restrict__ nbr, float* __restrict__ out) {
  int i = blockIdx.x*blockDim.x + threadIdx.x;
  if (i >= NN*128) return;
  int c = i & 31; int rr = i >> 5; int m = rr & 3; int n = rr >> 2;
  float s = 0.f;
  for (int k = 0; k < KK; ++k) {
    int src = nbr[n*KK + k];
    s += g[(size_t)(src*4 + m)*32 + c] * r[(size_t)(n*KK + k)*32 + c];
  }
  out[i] = s * (1.f/30.f);
}

__global__ void copy_kernel(const float* __restrict__ a, float* __restrict__ o, int count) {
  int i = blockIdx.x*blockDim.x + threadIdx.x;
  if (i < count) o[i] = a[i];
}

// x[n,m,:] = [node | res]; xin[n,:] = [node0 | res0 | seq]
__global__ void xcat_kernel(const float* __restrict__ node, const float* __restrict__ res,
                            const float* __restrict__ seq, float* __restrict__ x,
                            float* __restrict__ xin) {
  int i = blockIdx.x*blockDim.x + threadIdx.x;
  if (i >= NN*64) return;
  int c = i & 63; int n = i >> 6;
  for (int m = 0; m < 4; ++m)
    x[(size_t)(n*4+m)*64 + c] = (c < 32) ? node[(size_t)(n*4+m)*32 + c]
                                         : res[(size_t)(n*4+m)*32 + (c-32)];
  xin[(size_t)n*96 + c] = (c < 32) ? node[(size_t)n*128 + c] : res[(size_t)n*128 + (c-32)];
  if (c < 32) xin[(size_t)n*96 + 64 + c] = seq[(size_t)n*32 + c];
}

// h = relu(t + p1[nbr] + p2[n] + ba1), in place on t (N*K, 32)
__global__ void attn_h_kernel(float* __restrict__ t, const float* __restrict__ p1,
                              const float* __restrict__ p2, const float* __restrict__ ba1,
                              const int* __restrict__ nbr) {
  int i = blockIdx.x*blockDim.x + threadIdx.x;
  if (i >= NKE*32) return;
  int c = i & 31; int e = i >> 5; int n = e / KK; int src = nbr[e];
  float v = t[i] + p1[(size_t)src*32 + c] + p2[(size_t)n*32 + c] + ba1[c];
  t[i] = fmaxf(v, 0.f);
}

// softmax over K for each (n, head), in place on (N,K,8)
__global__ void softmax_kernel(float* __restrict__ lg) {
  int i = blockIdx.x*blockDim.x + threadIdx.x;
  if (i >= NN*8) return;
  int h = i & 7; int n = i >> 3;
  float m = -1e30f;
  for (int k = 0; k < KK; ++k) m = fmaxf(m, lg[(size_t)(n*KK+k)*8 + h]);
  float s = 0.f;
  for (int k = 0; k < KK; ++k) s += expf(lg[(size_t)(n*KK+k)*8 + h] - m);
  float inv = 1.f / s;
  for (int k = 0; k < KK; ++k) {
    size_t o = (size_t)(n*KK+k)*8 + h;
    lg[o] = expf(lg[o] - m) * inv;
  }
}

// node_pre[n,m,c] = sum_k attn[n,k,h(c)] * y[nbr[n,k],m,c]
__global__ void aggregate_kernel(const float* __restrict__ attn, const float* __restrict__ y,
                                 const int* __restrict__ nbr, float* __restrict__ np_) {
  int i = blockIdx.x*blockDim.x + threadIdx.x;
  if (i >= NN*512) return;
  int c = i & 127; int rr = i >> 7; int m = rr & 3; int n = rr >> 2; int h = c >> 4;
  float s = 0.f;
  for (int k = 0; k < KK; ++k) {
    int src = nbr[n*KK + k];
    s += attn[(size_t)(n*KK+k)*8 + h] * y[(size_t)(src*4 + m)*128 + c];
  }
  np_[i] = s;
}

// res = norm_so3(res + ru, g0, g1); wave32 per node, lane = channel
__global__ void res_norm_kernel(float* __restrict__ res, const float* __restrict__ ru,
                                const float* __restrict__ g0, const float* __restrict__ g1) {
  int n = blockIdx.x * (blockDim.x >> 5) + (threadIdx.x >> 5);
  if (n >= NN) return;
  int c = threadIdx.x & 31;
  size_t b = (size_t)n * 128;
  float v0 = res[b + c]      + (ru ? ru[b + c]      : 0.f);
  float v1 = res[b + 32 + c] + (ru ? ru[b + 32 + c] : 0.f);
  float v2 = res[b + 64 + c] + (ru ? ru[b + 64 + c] : 0.f);
  float v3 = res[b + 96 + c] + (ru ? ru[b + 96 + c] : 0.f);
  float s0 = wave_sum(v0*v0) * (1.f/32.f);
  float s1 = wave_sum(v1*v1 + v2*v2 + v3*v3) * (1.f/96.f);
  float i0 = 1.f / sqrtf(s0 + 1e-8f);
  float i1 = 1.f / sqrtf(s1 + 1e-8f);
  res[b + c]      = v0 * i0 * g0[c];
  res[b + 32 + c] = v1 * i1 * g1[c];
  res[b + 64 + c] = v2 * i1 * g1[c];
  res[b + 96 + c] = v3 * i1 * g1[c];
}

// out = layer_norm(a (+ b)); wave32 per node over 32 channels
__global__ void ln_kernel(const float* __restrict__ a, const float* __restrict__ badd,
                          float* __restrict__ out, const float* __restrict__ g,
                          const float* __restrict__ bet) {
  int n = blockIdx.x * (blockDim.x >> 5) + (threadIdx.x >> 5);
  if (n >= NN) return;
  int c = threadIdx.x & 31;
  float v = a[(size_t)n*32 + c] + (badd ? badd[(size_t)n*32 + c] : 0.f);
  float mean = wave_sum(v) * (1.f/32.f);
  float d = v - mean;
  float var = wave_sum(d*d) * (1.f/32.f);
  out[(size_t)n*32 + c] = d * rsqrtf(var + 1e-5f) * g[c] + bet[c];
}

// h2 = relu(t2 + q1[nbr] + q2[n] + be1), in place on t2 (N*K, 64)
__global__ void edge_h_kernel(float* __restrict__ t2, const float* __restrict__ q1,
                              const float* __restrict__ q2, const float* __restrict__ be1,
                              const int* __restrict__ nbr) {
  int i = blockIdx.x*blockDim.x + threadIdx.x;
  if (i >= NKE*64) return;
  int c = i & 63; int e = i >> 6; int n = e / KK; int src = nbr[e];
  float v = t2[i] + q1[(size_t)src*64 + c] + q2[(size_t)n*64 + c] + be1[c];
  t2[i] = fmaxf(v, 0.f);
}

// latent[n,a,d] = mean_k ag[nbr[n,k], 1+d, a] * r2[n,k,a]
__global__ void atoms_reduce_kernel(const float* __restrict__ ag, const float* __restrict__ r2,
                                    const int* __restrict__ nbr, float* __restrict__ outL) {
  int i = blockIdx.x*blockDim.x + threadIdx.x;
  if (i >= NN*3*91) return;
  int a = i % 91; int t = i / 91; int d = t % 3; int n = t / 3; int m = d + 1;
  float s = 0.f;
  for (int k = 0; k < KK; ++k) {
    int src = nbr[n*KK + k];
    s += ag[(size_t)(src*4 + m)*91 + a] * r2[(size_t)(n*KK + k)*91 + a];
  }
  outL[(size_t)n*273 + a*3 + d] = s * (1.f/30.f);
}

__global__ void logsoftmax_kernel(float* __restrict__ L) {
  int n = blockIdx.x*blockDim.x + threadIdx.x;
  if (n >= NN) return;
  float m = -1e30f;
  for (int j = 0; j < 20; ++j) m = fmaxf(m, L[(size_t)n*20 + j]);
  float s = 0.f;
  for (int j = 0; j < 20; ++j) s += expf(L[(size_t)n*20 + j] - m);
  float lse = logf(s);
  for (int j = 0; j < 20; ++j) L[(size_t)n*20 + j] = L[(size_t)n*20 + j] - m - lse;
}

// ---------------------------------------------------------------------------
extern "C" void kernel_launch(void* const* d_in, const int* in_sizes, int n_in,
                              void* d_out, int out_size, void* d_ws, size_t ws_size,
                              hipStream_t stream) {
  (void)in_sizes; (void)n_in; (void)out_size; (void)ws_size;
  const float* bb            = (const float*)d_in[0];
  const unsigned char* xmask = (const unsigned char*)d_in[1];
  const int* batch           = (const int*)d_in[2];
  const float* res_emb       = (const float*)d_in[3];
  const float* seqf          = (const float*)d_in[4];
  auto F = [&](int i){ return (const float*)d_in[i]; };
  int p = 5;
  const float *eW0=F(p+0),*eW1=F(p+1),*eWr1=F(p+2),*ebr1=F(p+3),*eWr2=F(p+4),*ebr2=F(p+5); p+=6;
  const float *aW0=F(p+0),*aW1=F(p+1),*aWr1=F(p+2),*abr1=F(p+3),*aWr2=F(p+4),*abr2=F(p+5); p+=6;
  (void)aW0;
  struct LP { const float *Wf,*bf,*Wa1,*ba1,*Wa2,*ba2,*Wv,*Wo,*Wr0,*br0,*Wr1l,*g0,*g1,*Ws,*bs,*lg,*lb,*We1,*be1,*We2,*be2; } L[4];
  for (int l = 0; l < 4; ++l) {
    L[l].Wf=F(p+0);  L[l].bf=F(p+1);  L[l].Wa1=F(p+2);  L[l].ba1=F(p+3);
    L[l].Wa2=F(p+4); L[l].ba2=F(p+5); L[l].Wv=F(p+6);   L[l].Wo=F(p+7);
    L[l].Wr0=F(p+8); L[l].br0=F(p+9); L[l].Wr1l=F(p+10);
    L[l].g0=F(p+11); L[l].g1=F(p+12); L[l].Ws=F(p+13);  L[l].bs=F(p+14);
    L[l].lg=F(p+15); L[l].lb=F(p+16); L[l].We1=F(p+17); L[l].be1=F(p+18);
    L[l].We2=F(p+19); L[l].be2=F(p+20);
    p += 21;
  }
  const float* fg0 = F(p++); const float* fg1 = F(p++);
  const float *hg=F(p+0),*hb=F(p+1),*hW1=F(p+2),*hb1=F(p+3),*hW2=F(p+4),*hb2=F(p+5),*hW3=F(p+6),*hb3=F(p+7);

  // workspace layout (floats)
  float* w = (float*)d_ws;
  size_t o = 0;
  int*   nbr  = (int*)(w + o); o += NKE;
  float* edge = w + o; o += (size_t)NKE*32;
  float* big  = w + o; o += (size_t)NKE*128;
  float* x    = w + o; o += (size_t)NN*256;
  float* y    = w + o; o += (size_t)NN*512;
  float* node = w + o; o += (size_t)NN*128;
  float* res  = w + o; o += (size_t)NN*128;
  float* seq  = w + o; o += (size_t)NN*32;
  float* xin  = w + o; o += (size_t)NN*96;
  float* p1b  = w + o; o += (size_t)NN*32;
  float* p2b  = w + o; o += (size_t)NN*32;
  float* q1b  = w + o; o += (size_t)NN*64;
  float* q2b  = w + o; o += (size_t)NN*64;
  float* tmps = w + o; o += (size_t)NN*32;
  float* hn   = w + o; o += (size_t)NN*32;
  float* h1b  = w + o; o += (size_t)NN*64;
  float* h2b  = w + o; o += (size_t)NN*32;
  float* bbe  = w + o; o += (size_t)NN*28;
  float* gbuf = w + o; o += (size_t)NN*128;
  _Float16* wt = (_Float16*)(w + o); o += 16384;  // 32K halves staging for BT
  float* tbuf = big;                    // NK*32 scratch
  float* rbuf = big + (size_t)NKE*32;   // r / attn / r2 overlay
  float* npre = big + (size_t)NKE*40;   // 4N*128
  float* rub  = big + (size_t)NKE*60;   // 4N*32
  float* atomg = y;                     // reuse y after layers

  float* latent = (float*)d_out;
  float* logits = (float*)d_out + (size_t)NN*273;

  auto B1 = [](long total){ return (unsigned)((total + 255) / 256); };

  // weight-transpose then WMMA GEMM (stream-serialized on shared BT staging)
  auto gemm = [&](const float* A, int lda, const float* B, int ldb,
                  float* C, int ldc, int M, int Ka, int Nb,
                  const float* bias, int flags) {
    int Npad = ((Nb + 15) / 16) * 16;
    int tot = Npad * Ka;
    wtr_kernel<<<B1(tot), 256, 0, stream>>>(B, ldb, Ka, Nb, Npad, wt);
    long tiles = (long)(M / 16) * (Npad / 16);
    wmma_gemm_kernel<<<(int)((tiles + 7) / 8), 256, 0, stream>>>(
        A, lda, wt, C, ldc, M, Ka, Nb, bias, flags);
  };

  // ---- geometry ----
  knn_kernel<<<NN/KW, KW*32, 0, stream>>>(bb, batch, xmask, nbr);
  edge_feat_kernel<<<B1(NKE), 256, 0, stream>>>(bb, nbr, edge);
  bbemb_kernel<<<B1(NN), 256, 0, stream>>>(bb, bbe);
  gemb_kernel<<<B1((long)NN*128), 256, 0, stream>>>(bbe, eW0, eW1, gbuf);

  // ---- embed projection: r = relu(edge@Wr1+br1)@Wr2+br2; node = mean_k g[nbr]*r ----
  gemm(edge, 32, eWr1, 32, tbuf, 32, NKE, 32, 32, ebr1, GF_RELU);
  gemm(tbuf, 32, eWr2, 32, rbuf, 32, NKE, 32, 32, ebr2, 0);
  proj_reduce_kernel<<<B1((long)NN*128), 256, 0, stream>>>(gbuf, rbuf, nbr, node);

  copy_kernel<<<B1((long)NN*128), 256, 0, stream>>>(res_emb, res, NN*128);
  copy_kernel<<<B1((long)NN*32), 256, 0, stream>>>(seqf, seq, NN*32);

  // ---- 4 message-passing layers ----
  for (int l = 0; l < 4; ++l) {
    const LP& q = L[l];
    xcat_kernel<<<B1((long)NN*64), 256, 0, stream>>>(node, res, seq, x, xin);
    gemm(xin, 96, q.Wf, 64, x, 256, NN, 96, 64, q.bf, 0);          // x[:,0,:]
    // attention
    gemm(x, 256, q.Wa1,           32, p1b, 32, NN, 64, 32, nullptr, 0);
    gemm(x, 256, q.Wa1 + 64*32,   32, p2b, 32, NN, 64, 32, nullptr, 0);
    gemm(edge, 32, q.Wa1 + 128*32, 32, tbuf, 32, NKE, 32, 32, nullptr, 0);
    attn_h_kernel<<<B1((long)NKE*32), 256, 0, stream>>>(tbuf, p1b, p2b, q.ba1, nbr);
    gemm(tbuf, 32, q.Wa2, 8, rbuf, 8, NKE, 32, 8, q.ba2, 0);
    softmax_kernel<<<B1((long)NN*8), 256, 0, stream>>>(rbuf);
    // value path (rotations cancel): y = x @ Wv per node, gather-aggregate, @ Wo
    gemm(x, 64, q.Wv, 128, y, 128, NN*4, 64, 128, nullptr, 0);
    aggregate_kernel<<<B1((long)NN*512), 256, 0, stream>>>(rbuf, y, nbr, npre);
    gemm(npre, 128, q.Wo, 32, node, 32, NN*4, 128, 32, nullptr, 0);
    // residual SO(3) update
    gemm(node, 32,  q.Wr1l, 32, rub, 32,  NN*4, 32, 32, nullptr, 0);
    gemm(node, 128, q.Wr0,  32, rub, 128, NN,   32, 32, q.br0, 0);
    res_norm_kernel<<<NN/8, 256, 0, stream>>>(res, rub, q.g0, q.g1);
    // sequence update
    gemm(node, 128, q.Ws, 32, tmps, 32, NN, 32, 32, q.bs, 0);
    ln_kernel<<<NN/8, 256, 0, stream>>>(seq, tmps, seq, q.lg, q.lb);
    // edge update MLP (fused residual add in GEMM epilogue)
    gemm(node, 128, q.We1,         64, q1b, 64, NN, 32, 64, nullptr, 0);
    gemm(node, 128, q.We1 + 32*64, 64, q2b, 64, NN, 32, 64, nullptr, 0);
    gemm(edge, 32,  q.We1 + 64*64, 64, big, 64, NKE, 32, 64, nullptr, 0);
    edge_h_kernel<<<B1((long)NKE*64), 256, 0, stream>>>(big, q1b, q2b, q.be1, nbr);
    gemm(big, 64, q.We2, 32, edge, 32, NKE, 64, 32, q.be2, GF_ACC);
  }

  // ---- final norm + atoms projection -> latent ----
  res_norm_kernel<<<NN/8, 256, 0, stream>>>(res, nullptr, fg0, fg1);
  gemm(edge, 32, aWr1, 32, tbuf, 32, NKE, 32, 32, abr1, GF_RELU);
  gemm(tbuf, 32, aWr2, 91, rbuf, 91, NKE, 32, 91, abr2, 0);
  gemm(res, 32, aW1, 91, atomg, 91, NN*4, 32, 91, nullptr, 0);
  atoms_reduce_kernel<<<B1((long)NN*3*91), 256, 0, stream>>>(atomg, rbuf, nbr, latent);

  // ---- sequence head -> log-softmax logits ----
  ln_kernel<<<NN/8, 256, 0, stream>>>(seqf, nullptr, hn, hg, hb);
  gemm(hn,  32, hW1, 64, h1b, 64, NN, 32, 64, hb1, GF_RELU);
  gemm(h1b, 64, hW2, 32, h2b, 32, NN, 64, 32, hb2, GF_RELU);
  gemm(h2b, 32, hW3, 20, logits, 20, NN, 32, 20, hb3, 0);
  logsoftmax_kernel<<<B1(NN), 256, 0, stream>>>(logits);
}
